// TimeSeriesEncoderMamba_12446815224154
// MI455X (gfx1250) — compile-verified, matrix operand order checked by reading, not run on hardware
//
#include <hip/hip_runtime.h>
#include <hip/hip_bf16.h>
#include <math.h>

// ---------------- WMMA types ----------------
typedef __bf16 v16bf __attribute__((ext_vector_type(16)));
typedef float  v8f   __attribute__((ext_vector_type(8)));

// Load 32 consecutive-K fp32 values (per ISA 16-bit A/B fragment layout) for
// this lane, convert to bf16 fragment.
//   lane m = lane&15 owns row/col m; khalf = lane>>4
//   element j -> k = (j>=8 ? 16 : 0) + khalf*8 + (j&7)
__device__ __forceinline__ v16bf load_frag(const float* rowp, int k0, int khalf) {
    const float4* p0 = (const float4*)(rowp + k0 + khalf * 8);
    const float4* p1 = (const float4*)(rowp + k0 + 16 + khalf * 8);
    float4 a0 = p0[0], a1 = p0[1];
    float4 b0 = p1[0], b1 = p1[1];
    v16bf r;
    r[0]  = (__bf16)a0.x; r[1]  = (__bf16)a0.y; r[2]  = (__bf16)a0.z; r[3]  = (__bf16)a0.w;
    r[4]  = (__bf16)a1.x; r[5]  = (__bf16)a1.y; r[6]  = (__bf16)a1.z; r[7]  = (__bf16)a1.w;
    r[8]  = (__bf16)b0.x; r[9]  = (__bf16)b0.y; r[10] = (__bf16)b0.z; r[11] = (__bf16)b0.w;
    r[12] = (__bf16)b1.x; r[13] = (__bf16)b1.y; r[14] = (__bf16)b1.z; r[15] = (__bf16)b1.w;
    return r;
}

#define WMMA_BF16(a, b, c) \
    __builtin_amdgcn_wmma_f32_16x16x32_bf16(false, (a), false, (b), (short)0, (c), false, false)

// C[M,N] = A[M,K] * W[N,K]^T  (+bias, or += into C). bf16 WMMA, fp32 acc.
// Block = 128 threads = 4 waves; each wave -> 32(M) x 64(N) tile (8 wmma/K-step).
// Grid = (N/64, M/128). M mult of 128, N mult of 64 (min 64), K mult of 32.
template<bool ACC, bool BIAS>
__global__ void gemm_wmma(const float* __restrict__ A, int lda,
                          const float* __restrict__ W,
                          float* __restrict__ C, int ldc,
                          int M, int N, int K, const float* __restrict__ bias) {
    const int lane  = threadIdx.x & 31;
    const int wm    = threadIdx.x >> 5;            // wave index in block (M dir)
    const int m     = lane & 15;
    const int khalf = lane >> 4;
    const int mbase = blockIdx.y * 128 + wm * 32;
    const int n0    = blockIdx.x * 64;

    const float* arow0 = A + (size_t)(mbase + m) * lda;
    const float* arow1 = A + (size_t)(mbase + 16 + m) * lda;
    const float* br0   = W + (size_t)(n0 + m) * K;
    const float* br1   = br0 + (size_t)16 * K;
    const float* br2   = br0 + (size_t)32 * K;
    const float* br3   = br0 + (size_t)48 * K;

    v8f a00 = {}, a01 = {}, a02 = {}, a03 = {};
    v8f a10 = {}, a11 = {}, a12 = {}, a13 = {};

    for (int k0 = 0; k0 < K; k0 += 32) {
        // keep the fp32 weight/activation stream ahead in cache
        __builtin_prefetch(arow0 + k0 + 64, 0, 1);
        __builtin_prefetch(br0 + k0 + 64, 0, 1);
        __builtin_prefetch(br2 + k0 + 64, 0, 1);

        v16bf af0 = load_frag(arow0, k0, khalf);
        v16bf af1 = load_frag(arow1, k0, khalf);
        v16bf b0  = load_frag(br0, k0, khalf);
        a00 = WMMA_BF16(af0, b0, a00);
        a10 = WMMA_BF16(af1, b0, a10);
        v16bf b1  = load_frag(br1, k0, khalf);
        a01 = WMMA_BF16(af0, b1, a01);
        a11 = WMMA_BF16(af1, b1, a11);
        v16bf b2  = load_frag(br2, k0, khalf);
        a02 = WMMA_BF16(af0, b2, a02);
        a12 = WMMA_BF16(af1, b2, a12);
        v16bf b3  = load_frag(br3, k0, khalf);
        a03 = WMMA_BF16(af0, b3, a03);
        a13 = WMMA_BF16(af1, b3, a13);
    }

    // Epilogue: VGPR r, lane -> row = mbase + ms*16 + khalf*8 + r,
    //                           col = n0 + t*16 + m
    v8f accs[2][4] = {{a00, a01, a02, a03}, {a10, a11, a12, a13}};
#pragma unroll
    for (int ms = 0; ms < 2; ++ms) {
#pragma unroll
        for (int t = 0; t < 4; ++t) {
            int col = n0 + t * 16 + m;
            float bv = BIAS ? bias[col] : 0.0f;
#pragma unroll
            for (int r = 0; r < 8; ++r) {
                int row = mbase + ms * 16 + khalf * 8 + r;
                float v = accs[ms][t][r] + bv;
                size_t off = (size_t)row * ldc + col;
                if (ACC) v += C[off];
                C[off] = v;
            }
        }
    }
}

// ---------------- patch embed: x[b,p,:] = ts[b,4p:4p+4] @ emb_w^T + emb_b ----
__global__ void embed_kernel(const float* __restrict__ ts,
                             const float* __restrict__ ew,
                             const float* __restrict__ eb,
                             float* __restrict__ x) {
    int idx = blockIdx.x * 256 + threadIdx.x;      // 1024 rows * 512
    int dch = idx & 511;
    int r   = idx >> 9;                            // b*256 + p
    int b   = r >> 8, p = r & 255;
    const float* t = ts + b * 1024 + p * 4;
    float acc = eb[dch];
#pragma unroll
    for (int k = 0; k < 4; ++k) acc += t[k] * ew[dch * 4 + k];
    x[idx] = acc;
}

// ---------------- RMSNorm over D_MODEL=512, one block per row ---------------
__global__ void rmsnorm_kernel(const float* __restrict__ x,
                               const float* __restrict__ w,
                               float* __restrict__ xn) {
    __shared__ float red[256];
    int row = blockIdx.x, tid = threadIdx.x;
    const float* xr = x + (size_t)row * 512;
    float v0 = xr[tid], v1 = xr[tid + 256];
    red[tid] = v0 * v0 + v1 * v1;
    __syncthreads();
    for (int s = 128; s > 0; s >>= 1) {
        if (tid < s) red[tid] += red[tid + s];
        __syncthreads();
    }
    float scale = rsqrtf(red[0] * (1.0f / 512.0f) + 1e-5f);
    xn[(size_t)row * 512 + tid]       = v0 * scale * w[tid];
    xn[(size_t)row * 512 + tid + 256] = v1 * scale * w[tid + 256];
}

// ---- causal depthwise conv1d(k=4) + bias + SiLU, both directions -----------
// u = xz2[:, d*2048 + 0:1024] at original positions; uc written in scan order.
__global__ void conv_silu_kernel(const float* __restrict__ xz2,
                                 const float* __restrict__ cw,   // layer base (2,1024,4)
                                 const float* __restrict__ cb,   // layer base (2,1024)
                                 float* __restrict__ uc) {
    int idx = blockIdx.x * 256 + threadIdx.x;      // ((b*2+d)*256+l)*1024 + e
    int e  = idx & 1023;
    int r  = idx >> 10;
    int l  = r & 255;
    int r2 = r >> 8;
    int d  = r2 & 1;
    int b  = r2 >> 1;
    float acc = cb[d * 1024 + e];
#pragma unroll
    for (int j = 0; j < 4; ++j) {
        int ls = l - 3 + j;                        // scan-order source pos
        if (ls >= 0) {
            int lo = d ? 255 - ls : ls;            // original pos
            acc += xz2[((size_t)((b << 8) + lo)) * 4096 + d * 2048 + e]
                   * cw[d * 4096 + e * 4 + j];
        }
    }
    uc[((size_t)((b << 8) + l)) * 2048 + d * 1024 + e] = acc / (1.0f + __expf(-acc));
}

// ---- selective scan, both directions in one launch -------------------------
// h[n] = exp(dt*A[n])*h[n] + dt*u*B[n]; y = h.C ; fuses softplus(dt_proj+dt_b),
// +u*Dp, *silu(z), and the un-flip for d=1. Grid = 64 blocks (b,d,chunk).
__global__ void scan_kernel(const float* __restrict__ deltp,   // (1024, 2048)
                            const float* __restrict__ uc,      // (1024, 2048)
                            const float* __restrict__ xdbl,    // (1024, 128)
                            const float* __restrict__ xz2,     // (1024, 4096)
                            const float* __restrict__ A_log_l, // layer base (2,1024,16)
                            const float* __restrict__ dt_b_l,  // layer base (2,1024)
                            const float* __restrict__ Dp_l,    // layer base (2,1024)
                            float* __restrict__ yfull) {       // (1024, 2048)
    __shared__ float sB[16], sC[16];
    int bi    = blockIdx.x;
    int b     = bi >> 4;
    int d     = (bi >> 3) & 1;
    int chunk = bi & 7;
    int e     = (chunk << 7) + threadIdx.x;        // 128 threads -> 128 channels
    float Ae[16], h[16];
#pragma unroll
    for (int n = 0; n < 16; ++n) {
        Ae[n] = -__expf(A_log_l[(size_t)d * 16384 + e * 16 + n]);
        h[n]  = 0.0f;
    }
    float dtb = dt_b_l[d * 1024 + e], dpe = Dp_l[d * 1024 + e];
    for (int l = 0; l < 256; ++l) {
        if (threadIdx.x < 32) {
            float v = xdbl[((size_t)((b << 8) + l)) * 128 + d * 64 + 32 + threadIdx.x];
            if (threadIdx.x < 16) sB[threadIdx.x] = v; else sC[threadIdx.x - 16] = v;
        }
        __syncthreads();
        size_t row  = (size_t)((b << 8) + l);
        float dp    = deltp[row * 2048 + d * 1024 + e] + dtb;
        float delta = dp > 20.0f ? dp : log1pf(__expf(dp));
        float uv    = uc[row * 2048 + d * 1024 + e];
        float du    = delta * uv;
        float y = 0.0f;
#pragma unroll
        for (int n = 0; n < 16; ++n) {
            h[n] = __expf(delta * Ae[n]) * h[n] + du * sB[n];
            y += h[n] * sC[n];
        }
        int lo = d ? 255 - l : l;                  // original position
        size_t rowo = (size_t)((b << 8) + lo);
        float z  = xz2[rowo * 4096 + d * 2048 + 1024 + e];
        float sz = z / (1.0f + __expf(-z));
        yfull[rowo * 2048 + d * 1024 + e] = (y + uv * dpe) * sz;
        __syncthreads();
    }
}

// ---------------- host side -------------------------------------------------
static void run_gemm(const float* A, int lda, const float* W, float* C, int ldc,
                     int M, int N, int K, const float* bias, bool accum,
                     hipStream_t s) {
    dim3 g(N / 64, M / 128), blk(128);
    if (accum)      gemm_wmma<true,  false><<<g, blk, 0, s>>>(A, lda, W, C, ldc, M, N, K, nullptr);
    else if (bias)  gemm_wmma<false, true ><<<g, blk, 0, s>>>(A, lda, W, C, ldc, M, N, K, bias);
    else            gemm_wmma<false, false><<<g, blk, 0, s>>>(A, lda, W, C, ldc, M, N, K, nullptr);
}

extern "C" void kernel_launch(void* const* d_in, const int* in_sizes, int n_in,
                              void* d_out, int out_size, void* d_ws, size_t ws_size,
                              hipStream_t stream) {
    (void)in_sizes; (void)n_in; (void)out_size; (void)ws_size;
    const float* ts      = (const float*)d_in[0];
    /* d_in[1] = mask (unused by reference) */
    const float* emb_w   = (const float*)d_in[2];
    const float* emb_b   = (const float*)d_in[3];
    const float* norm_w  = (const float*)d_in[4];
    const float* in_w    = (const float*)d_in[5];
    const float* conv_w  = (const float*)d_in[6];
    const float* conv_b  = (const float*)d_in[7];
    const float* xproj_w = (const float*)d_in[8];
    const float* dt_w    = (const float*)d_in[9];
    const float* dt_b    = (const float*)d_in[10];
    const float* A_log   = (const float*)d_in[11];
    const float* Dpv     = (const float*)d_in[12];
    const float* out_w   = (const float*)d_in[13];
    const float* proj_w  = (const float*)d_in[14];
    const float* proj_b  = (const float*)d_in[15];
    float* out = (float*)d_out;

    // workspace partition (floats); rows M = B*L = 1024, dirs packed in columns
    float* ws    = (float*)d_ws;
    float* x     = ws;                     // 1024 x 512
    float* xn    = x     + 1024 * 512;     // 1024 x 512
    float* xz2   = xn    + 1024 * 512;     // 1024 x 4096  (dir0: u|z, dir1: u|z)
    float* ucb   = xz2   + 1024 * 4096;    // 1024 x 2048  (scan order)
    float* xdbl  = ucb   + 1024 * 2048;    // 1024 x 128   (dt|B|C per dir)
    float* deltp = xdbl  + 1024 * 128;     // 1024 x 2048
    float* yfull = deltp + 1024 * 2048;    // 1024 x 2048

    embed_kernel<<<2048, 256, 0, stream>>>(ts, emb_w, emb_b, x);

    for (int L = 0; L < 8; ++L) {
        rmsnorm_kernel<<<1024, 256, 0, stream>>>(x, norm_w + L * 512, xn);

        // in_proj, BOTH directions fused: (1024x512) @ (4096x512)^T -> xz2
        run_gemm(xn, 512, in_w + (size_t)L * 2 * 2048 * 512, xz2, 4096,
                 1024, 4096, 512, nullptr, false, stream);

        // depthwise conv + SiLU, both directions (scan order)
        conv_silu_kernel<<<8192, 256, 0, stream>>>(
            xz2, conv_w + (size_t)L * 2 * 1024 * 4, conv_b + (size_t)L * 2 * 1024, ucb);

        for (int d = 0; d < 2; ++d) {
            size_t ld = (size_t)(L * 2 + d);
            // x_proj: (1024x1024) @ (64x1024)^T -> xdbl cols [64d,64d+64)
            run_gemm(ucb + d * 1024, 2048, xproj_w + ld * 64 * 1024,
                     xdbl + d * 64, 128, 1024, 64, 1024, nullptr, false, stream);
            // dt_proj: (1024x32) @ (1024x32)^T -> deltp cols [1024d, ...)
            run_gemm(xdbl + d * 64, 128, dt_w + ld * 1024 * 32,
                     deltp + d * 1024, 2048, 1024, 1024, 32, nullptr, false, stream);
        }

        // selective scan, both directions concurrently (64 blocks)
        scan_kernel<<<64, 128, 0, stream>>>(
            deltp, ucb, xdbl, xz2,
            A_log + (size_t)L * 2 * 1024 * 16,
            dt_b + (size_t)L * 2 * 1024,
            Dpv + (size_t)L * 2 * 1024, yfull);

        for (int d = 0; d < 2; ++d) {
            size_t ld = (size_t)(L * 2 + d);
            // out_proj accumulated into residual x: x += yfull_d @ out_w^T
            run_gemm(yfull + d * 1024, 2048, out_w + ld * 512 * 1024,
                     x, 512, 1024, 512, 1024, nullptr, true, stream);
        }
    }
    // final projection straight into d_out (flat layout == reference reshape)
    run_gemm(x, 512, proj_w, out, 1024, 1024, 1024, 512, proj_b, false, stream);
}
